// AttentionHead_82557861364002
// MI455X (gfx1250) — compile-verified, matrix operand order checked by reading, not run on hardware
//
#include <hip/hip_runtime.h>
#include <hip/hip_bf16.h>

// ---------------------------------------------------------------------------
// Causal MHA for MI455X (gfx1250, wave32, WMMA).
//   B=2, T=2048, C=1024, H=16, D=64.
// Pipeline:
//   1) cvt:      X f32 -> f16 (row-major [B*T, C])
//   2) wtrans:   W f32 [k][n] -> f16 Wt [n][k]  (x4 weights)
//   3) gemm<0>:  Q = Xh*Wq + bq  -> f16 [B,H,T,D]
//      gemm<0>:  K = Xh*Wk + bk  -> f16 [B,H,T,D]
//      gemm<1>:  V = Xh*Wv + bv  -> f16 [B,H,D,T]   (transposed for V^T x P^T)
//   4) flash:    per (b,h,32-query tile): S^T = K*Q^T, online softmax,
//                O^T += V^T * P^T    -> f16 [B,T,C]
//   5) gemm<2>:  out = Oh*Wo + bo -> f32 [B,T,C]
// ---------------------------------------------------------------------------

typedef _Float16 f16;
typedef __attribute__((ext_vector_type(16))) _Float16 v16h;
typedef __attribute__((ext_vector_type(8)))  _Float16 v8h;
typedef __attribute__((ext_vector_type(8)))  float    v8f;
typedef __attribute__((ext_vector_type(4)))  float    v4f;

static constexpr int Bb = 2, Tt = 2048, Cc = 1024, Hh = 16, Dd = 64;

__device__ __forceinline__ v16h cat8(v8h lo, v8h hi) {
  return __builtin_shufflevector(lo, hi, 0,1,2,3,4,5,6,7,8,9,10,11,12,13,14,15);
}

__device__ __forceinline__ v8f wmma_f16(v16h a, v16h b, v8f c) {
  // D(f32 16x16) = A(16x32 f16) * B(32x16 f16) + C
  return __builtin_amdgcn_wmma_f32_16x16x32_f16(
      /*neg_a=*/false, a, /*neg_b=*/false, b,
      /*c_mod=*/(short)0, c, /*reuse_a=*/false, /*reuse_b=*/false);
}

// A-fragment (16x32, f16) from row-major [m][k] storage with stride (halves).
// Lane layout (ISA 7.12.2): lane = 16*half + m ; element j -> K = (j>>3)*16 + half*8 + (j&7)
// => two contiguous 16B loads per lane. Identical pattern serves B-fragments
// when the operand is stored [n][k]-major.
template <typename P>
__device__ __forceinline__ v16h load_frag_rowmajor(const P* base, int row,
                                                   int stride, int half) {
  const f16* p = (const f16*)base + (size_t)row * stride;
  v8h lo = *(const v8h*)(p + half * 8);
  v8h hi = *(const v8h*)(p + 16 + half * 8);
  return cat8(lo, hi);
}

// ---------------------------------------------------------------------------
// Prep kernels
// ---------------------------------------------------------------------------
__global__ void cvt_f32_to_f16(const float* __restrict__ in,
                               f16* __restrict__ out, int n4) {
  int i = blockIdx.x * blockDim.x + threadIdx.x;
  if (i < n4) {
    v4f v = *(const v4f*)(in + i * 4);
    f16* o = out + i * 4;
    o[0] = (f16)v.x; o[1] = (f16)v.y; o[2] = (f16)v.z; o[3] = (f16)v.w;
  }
}

// W [k][n] f32 (1024x1024) -> Wt [n][k] f16. k fastest in output => coalesced writes.
__global__ void wtrans_f32_to_f16(const float* __restrict__ W,
                                  f16* __restrict__ Wt) {
  int t = blockIdx.x * blockDim.x + threadIdx.x;   // 0 .. 1M-1
  int k = t & (Cc - 1);
  int n = t >> 10;
  Wt[(size_t)n * Cc + k] = (f16)W[(size_t)k * Cc + n];
}

// ---------------------------------------------------------------------------
// WMMA GEMM:  Out[M x 1024] = A[M x 1024] * Bt^T + bias
//   A  : f16 row-major [m][k] ; Bt : f16 [n][k]-major (pre-transposed weights)
// Block 256 thr = 8 waves (4 x 2), block tile 128x128, wave tile 32x64.
// K-step 32, double-buffered LDS, global loads issued before compute so
// s_wait_loadcnt lands after the WMMA chain (ldg -> compute -> sts -> barrier).
// MODE 0: f16 out [B,H,T,D] ; MODE 1: f16 out [B,H,D,T] ; MODE 2: f32 out [m][n]
// ---------------------------------------------------------------------------
template <int MODE>
__global__ __launch_bounds__(256) void gemm_wmma(
    const f16* __restrict__ A, const f16* __restrict__ Bt,
    const float* __restrict__ bias, void* __restrict__ Out) {
  constexpr int K = Cc;
  constexpr int LDP = 40;                     // padded row stride (halves), 80B
  __shared__ f16 sA[2][128 * LDP];
  __shared__ f16 sB[2][128 * LDP];

  const int tid  = threadIdx.x;
  const int lane = tid & 31;
  const int wave = tid >> 5;
  const int wrow = wave & 3;                  // 4 wave-rows  -> M
  const int wcol = wave >> 2;                 // 2 wave-cols  -> N
  const int half = lane >> 4;
  const int l16  = lane & 15;
  const int m0 = blockIdx.x * 128;
  const int n0 = blockIdx.y * 128;

  v8f acc[2][4] = {};
  v8h rA[2], rB[2];                           // staging registers (16B x2 each)

  auto ldg = [&](int k0) {                    // issue global loads (no LDS yet)
#pragma unroll
    for (int i = 0; i < 2; ++i) {
      int c = tid * 2 + i, row = c >> 2, off = (c & 3) * 8;
      const f16* gA = A  + (size_t)(m0 + row) * K + k0 + off;
      const f16* gB = Bt + (size_t)(n0 + row) * K + k0 + off;
      rA[i] = *(const v8h*)gA;
      rB[i] = *(const v8h*)gB;
      if (k0 + 32 < K) {                      // gfx1250 global_prefetch_b8
        __builtin_prefetch(gA + 32, 0, 0);
        __builtin_prefetch(gB + 32, 0, 0);
      }
    }
  };
  auto sts = [&](int buf) {                   // commit staging regs to LDS
#pragma unroll
    for (int i = 0; i < 2; ++i) {
      int c = tid * 2 + i, row = c >> 2, off = (c & 3) * 8;
      *(v8h*)(sA[buf] + row * LDP + off) = rA[i];
      *(v8h*)(sB[buf] + row * LDP + off) = rB[i];
    }
  };

  ldg(0);
  sts(0);
  __syncthreads();

  int buf = 0;
  for (int k0 = 0; k0 < K; k0 += 32) {
    if (k0 + 32 < K) ldg(k0 + 32);            // in flight during compute

    v16h af[2], bf[4];
#pragma unroll
    for (int mt = 0; mt < 2; ++mt)
      af[mt] = load_frag_rowmajor(sA[buf], wrow * 32 + mt * 16 + l16, LDP, half);
#pragma unroll
    for (int nt = 0; nt < 4; ++nt)
      bf[nt] = load_frag_rowmajor(sB[buf], wcol * 64 + nt * 16 + l16, LDP, half);
#pragma unroll
    for (int mt = 0; mt < 2; ++mt)
#pragma unroll
      for (int nt = 0; nt < 4; ++nt)
        acc[mt][nt] = wmma_f16(af[mt], bf[nt], acc[mt][nt]);

    if (k0 + 32 < K) sts(buf ^ 1);            // wait_loadcnt happens here
    __syncthreads();
    buf ^= 1;
  }

  // ---- epilogue: C layout: VGPR r -> row m = r + 8*half, col n = l16 ----
#pragma unroll
  for (int mt = 0; mt < 2; ++mt)
#pragma unroll
    for (int nt = 0; nt < 4; ++nt) {
      const int n  = n0 + wcol * 64 + nt * 16 + l16;
      const float bv = bias[n];
#pragma unroll
      for (int r = 0; r < 8; ++r) {
        const int m = m0 + wrow * 32 + mt * 16 + r + 8 * half;
        const float v = acc[mt][nt][r] + bv;
        if (MODE == 2) {
          ((float*)Out)[(size_t)m * Cc + n] = v;
        } else {
          const int b = m >> 11, t = m & (Tt - 1);   // m = b*T + t
          const int h = n >> 6,  d = n & (Dd - 1);   // n = h*D + d
          if (MODE == 0)
            ((f16*)Out)[(((size_t)(b * Hh + h) * Tt + t) * Dd) + d] = (f16)v;
          else
            ((f16*)Out)[(((size_t)(b * Hh + h) * Dd + d) * Tt) + t] = (f16)v;
        }
      }
    }
}

// ---------------------------------------------------------------------------
// Flash attention, transposed formulation. One wave = 32 queries (2 chunks of
// 16); key blocks of 32. K- and V-fragments are shared across both query
// chunks (2x arithmetic intensity vs 16-query waves).
//   S^T(16k x 16q) = K(16k x D) * Q^T(D x 16q)
//   O^T(D x 16q)  += V^T(D x 32k) * P^T(32k x 16q)
// q0 is a multiple of 32 => full key blocks need NO mask code; exactly one
// masked diagonal block per wave. Softmax stats are per-query = per-lane
// column of the C layout -> lane-local except one shfl_xor(16).
// ---------------------------------------------------------------------------
__global__ __launch_bounds__(128) void flash_attn(
    const f16* __restrict__ Qh,   // [B,H,T,D]
    const f16* __restrict__ Kh,   // [B,H,T,D]
    const f16* __restrict__ Vt,   // [B,H,D,T]
    f16* __restrict__ Oh) {       // [B,T,C] = [B,T,H,D]
  const int bh = blockIdx.x;                   // b*H + h
  const int b  = bh >> 4, h = bh & (Hh - 1);
  const int wave = threadIdx.x >> 5;
  const int lane = threadIdx.x & 31;
  const int half = lane >> 4, l16 = lane & 15;
  const int q0 = (blockIdx.y * 4 + wave) * 32; // 32-query tile per wave

  const f16* Qp = Qh + ((size_t)bh * Tt + q0) * Dd;
  const f16* Kp = Kh + (size_t)bh * Tt * Dd;
  const f16* Vp = Vt + (size_t)bh * Dd * Tt;

  // Q as B-operand ([q][d]-major == [n][k]-major): 2 chunks x 2 d-fragments.
  v16h qf[2][2];
#pragma unroll
  for (int qc = 0; qc < 2; ++qc)
#pragma unroll
    for (int dt = 0; dt < 2; ++dt)
      qf[qc][dt] = load_frag_rowmajor(Qp + qc * 16 * Dd + dt * 32, l16, Dd, half);

  v8f o[2][4] = {};                            // [chunk][d-block] O^T tiles
  float m_run[2] = {-__builtin_inff(), -__builtin_inff()};
  float l_run[2] = {0.0f, 0.0f};
  const float scl2 = 0.125f * 1.44269504088896f;  // 1/sqrt(D) * log2(e)

  auto body = [&](int k0, bool masked) {
    // ---- K A-fragments, shared by both query chunks (8 b128 loads) ----
    v16h ka[2][2];
#pragma unroll
    for (int kt = 0; kt < 2; ++kt)
#pragma unroll
      for (int dt = 0; dt < 2; ++dt)
        ka[kt][dt] = load_frag_rowmajor(
            Kp + (size_t)(k0 + kt * 16 + l16) * Dd + dt * 32, 0, Dd, half);

    if (!masked) {                             // prefetch next key block
      __builtin_prefetch(Kp + (size_t)(k0 + 32 + lane) * Dd, 0, 0);
      __builtin_prefetch(Vp + (size_t)lane * Tt + k0 + 32, 0, 0);
      __builtin_prefetch(Vp + (size_t)(lane + 32) * Tt + k0 + 32, 0, 0);
    }

    // ---- scores: S^T tiles, 8 WMMAs ----
    v8f s[2][2];                               // [chunk][key-tile]
#pragma unroll
    for (int qc = 0; qc < 2; ++qc)
#pragma unroll
      for (int kt = 0; kt < 2; ++kt) {
        v8f z = {};
        z = wmma_f16(ka[kt][0], qf[qc][0], z);
        s[qc][kt] = wmma_f16(ka[kt][1], qf[qc][1], z);
      }

    // ---- V^T A-fragments, shared by both chunks (8 b128 loads) ----
    v16h va[4];
#pragma unroll
    for (int dt = 0; dt < 4; ++dt)
      va[dt] = load_frag_rowmajor(Vp + (size_t)(dt * 16 + l16) * Tt + k0, 0,
                                  Tt, half);

#pragma unroll
    for (int qc = 0; qc < 2; ++qc) {
      const int q = q0 + qc * 16 + l16;        // this lane's query (C column)
      float p[16];
      float mblk = -__builtin_inff();
#pragma unroll
      for (int kt = 0; kt < 2; ++kt)
#pragma unroll
        for (int r = 0; r < 8; ++r) {
          float v = s[qc][kt][r] * scl2;       // scaled, log2 domain
          if (masked) {
            const int key = k0 + kt * 16 + r + 8 * half;
            if (key > q) v = -__builtin_inff();
          }
          p[kt * 8 + r] = v;
          mblk = fmaxf(mblk, v);
        }
      mblk = fmaxf(mblk, __shfl_xor(mblk, 16, 32));  // merge key halves
      const float m_new = fmaxf(m_run[qc], mblk);
      const float corr  = __builtin_exp2f(m_run[qc] - m_new);
      m_run[qc] = m_new;
      l_run[qc] *= corr;
#pragma unroll
      for (int dt = 0; dt < 4; ++dt)
#pragma unroll
        for (int r = 0; r < 8; ++r) o[qc][dt][r] *= corr;

      float lblk = 0.0f;
      v16h pf;                                 // P^T B-fragment is lane-local:
#pragma unroll
      for (int j = 0; j < 16; ++j) {           // b[j] = s-tile[j<8?0:1].c[j&7]
        const float e = __builtin_exp2f(p[j] - m_new);
        lblk += e;
        pf[j] = (f16)e;
      }
      l_run[qc] += lblk + __shfl_xor(lblk, 16, 32);

      // ---- O^T += V^T * P^T : 4 WMMAs per chunk ----
#pragma unroll
      for (int dt = 0; dt < 4; ++dt)
        o[qc][dt] = wmma_f16(va[dt], pf, o[qc][dt]);
    }
  };

  for (int k0 = 0; k0 < q0; k0 += 32) body(k0, false);  // full blocks: no mask
  body(q0, true);                                       // single diagonal block

  // ---- finalize: lane's column = query; rows d = dt*16 + r + 8*half ----
#pragma unroll
  for (int qc = 0; qc < 2; ++qc) {
    const float inv_l = 1.0f / l_run[qc];
    f16* op = Oh + ((size_t)b * Tt + q0 + qc * 16 + l16) * Cc + h * Dd;
#pragma unroll
    for (int dt = 0; dt < 4; ++dt)
#pragma unroll
      for (int r = 0; r < 8; ++r)
        op[dt * 16 + r + 8 * half] = (f16)(o[qc][dt][r] * inv_l);
  }
}

// ---------------------------------------------------------------------------
// Launch
// ---------------------------------------------------------------------------
extern "C" void kernel_launch(void* const* d_in, const int* in_sizes, int n_in,
                              void* d_out, int out_size, void* d_ws, size_t ws_size,
                              hipStream_t stream) {
  const float* X  = (const float*)d_in[0];
  const float* Wq = (const float*)d_in[1];
  const float* bq = (const float*)d_in[2];
  const float* Wk = (const float*)d_in[3];
  const float* bk = (const float*)d_in[4];
  const float* Wv = (const float*)d_in[5];
  const float* bv = (const float*)d_in[6];
  const float* Wo = (const float*)d_in[7];
  const float* bo = (const float*)d_in[8];
  float* out = (float*)d_out;

  const size_t NX = (size_t)Bb * Tt * Cc;      // 4M elements
  const size_t NW = (size_t)Cc * Cc;           // 1M elements

  // Workspace carve-up (f16, 2B each): ~40 MB total.
  f16* Xh  = (f16*)d_ws;                       //  8 MB; reused as Oh later
  f16* Wqt = Xh  + NX;                         //  2 MB
  f16* Wkt = Wqt + NW;
  f16* Wvt = Wkt + NW;
  f16* Wot = Wvt + NW;
  f16* Qh  = Wot + NW;                         //  8 MB
  f16* Kh  = Qh  + NX;                         //  8 MB
  f16* Vt  = Kh  + NX;                         //  8 MB
  f16* Oh  = Xh;                               // alias: Xh dead after QKV GEMMs

  // 1) X -> f16
  cvt_f32_to_f16<<<dim3((unsigned)(NX / 4 / 256)), 256, 0, stream>>>(X, Xh, (int)(NX / 4));
  // 2) weights -> f16, transposed to [n][k]
  wtrans_f32_to_f16<<<dim3((unsigned)(NW / 256)), 256, 0, stream>>>(Wq, Wqt);
  wtrans_f32_to_f16<<<dim3((unsigned)(NW / 256)), 256, 0, stream>>>(Wk, Wkt);
  wtrans_f32_to_f16<<<dim3((unsigned)(NW / 256)), 256, 0, stream>>>(Wv, Wvt);
  wtrans_f32_to_f16<<<dim3((unsigned)(NW / 256)), 256, 0, stream>>>(Wo, Wot);

  // 3) QKV projections (M=4096, N=1024 -> grid 32 x 8)
  dim3 ggrid(32, 8);
  gemm_wmma<0><<<ggrid, 256, 0, stream>>>(Xh, Wqt, bq, (void*)Qh);
  gemm_wmma<0><<<ggrid, 256, 0, stream>>>(Xh, Wkt, bk, (void*)Kh);
  gemm_wmma<1><<<ggrid, 256, 0, stream>>>(Xh, Wvt, bv, (void*)Vt);

  // 4) attention: grid (B*H, T/(4*32)) blocks of 4 waves
  flash_attn<<<dim3(Bb * Hh, Tt / 128), 128, 0, stream>>>(Qh, Kh, Vt, Oh);

  // 5) output projection -> f32 d_out
  gemm_wmma<2><<<ggrid, 256, 0, stream>>>(Oh, Wot, bo, (void*)out);
}
